// DeepSet_89816356094668
// MI455X (gfx1250) — compile-verified, matrix operand order checked by reading, not run on hardware
//
#include <hip/hip_runtime.h>
#include <hip/hip_bf16.h>

typedef __attribute__((ext_vector_type(16))) _Float16 v16h;
typedef __attribute__((ext_vector_type(8)))  _Float16 v8h;
typedef __attribute__((ext_vector_type(8)))  float    v8f;

#define BATCH   16384
#define XPITCH  544          // (2+32)*16 floats per input row
// workspace layout, offsets in halves (_Float16)
#define W0T_OFF   0          // [128][32]  phi_w0^T, K padded 16->32 with zeros
#define W1T_OFF   4096       // [128][128] phi_w1^T
#define W2T_OFF   20480      // [64][128]  phi_w2^T
#define RW0T_OFF  28672      // [256][96]  rho_w0^T
#define RW1T_OFF  53248      // [16][256]  rho_w1^T
#define XBUF_OFF  57344      // [16384][96] f16 rho input rows

__device__ __forceinline__ v8f wmma16(v16h a, v16h b, v8f c) {
    return __builtin_amdgcn_wmma_f32_16x16x32_f16(false, a, false, b, (short)0, c,
                                                  false, false);
}

__device__ __forceinline__ v16h cmb(v8h lo, v8h hi) {
    v16h r;
#pragma unroll
    for (int e = 0; e < 8; ++e) { r[e] = lo[e]; r[e + 8] = hi[e]; }
    return r;
}

// ---------------------------------------------------------------- prep ------
__global__ void ds_prep(const float* __restrict__ w0, const float* __restrict__ w1,
                        const float* __restrict__ w2, const float* __restrict__ rw0,
                        const float* __restrict__ rw1, _Float16* __restrict__ ws) {
    int i = blockIdx.x * blockDim.x + threadIdx.x;
    if (i < 4096) {                       // w0T [128][32], K padded
        int n = i >> 5, k = i & 31;
        ws[W0T_OFF + i] = (k < 16) ? (_Float16)w0[k * 128 + n] : (_Float16)0.f;
        return;
    }
    i -= 4096;
    if (i < 16384) {                      // w1T [128][128]
        int n = i >> 7, k = i & 127;
        ws[W1T_OFF + n * 128 + k] = (_Float16)w1[k * 128 + n];
        return;
    }
    i -= 16384;
    if (i < 8192) {                       // w2T [64][128]
        int n = i >> 7, k = i & 127;
        ws[W2T_OFF + n * 128 + k] = (_Float16)w2[k * 64 + n];
        return;
    }
    i -= 8192;
    if (i < 24576) {                      // rw0T [256][96]
        int n = i / 96, k = i % 96;
        ws[RW0T_OFF + n * 96 + k] = (_Float16)rw0[k * 256 + n];
        return;
    }
    i -= 24576;
    if (i < 4096) {                       // rw1T [16][256]
        int n = i >> 8, k = i & 255;
        ws[RW1T_OFF + n * 256 + k] = (_Float16)rw1[k * 16 + n];
    }
}

// ------------------------------------------------------------- phi + sum ----
#define PHI_WAVES 4
__global__ void ds_phi(const float* __restrict__ x, const float* __restrict__ b0,
                       const float* __restrict__ b1, const float* __restrict__ b2,
                       const _Float16* __restrict__ ws, _Float16* __restrict__ X) {
    __shared__ _Float16 bufA[PHI_WAVES][16 * 128];
    __shared__ _Float16 bufB[PHI_WAVES][16 * 128];
    const int lane = threadIdx.x & 31;
    const int wv   = threadIdx.x >> 5;
    const int wave = blockIdx.x * PHI_WAVES + wv;
    const int nW   = gridDim.x * PHI_WAVES;
    const _Float16* w0T = ws + W0T_OFF;
    const _Float16* w1T = ws + W1T_OFF;
    const _Float16* w2T = ws + W2T_OFF;
    const int nn = lane & 15;             // N (or M for A-operand) index
    const int kb = (lane >> 4) * 8;       // K base for this half-wave
    const int mb = (lane >> 4) * 8;       // M base for C-operand rows
    _Float16* A_ = bufA[wv];
    _Float16* B_ = bufB[wv];

    // per-lane loop-invariant biases (this lane always handles column nn of
    // each 16-wide N tile)
    float bias0[8], bias1[8], bias2x32[4];
#pragma unroll
    for (int n = 0; n < 8; ++n) bias0[n] = b0[n * 16 + nn];
#pragma unroll
    for (int n = 0; n < 8; ++n) bias1[n] = b1[n * 16 + nn];
#pragma unroll
    for (int n = 0; n < 4; ++n) bias2x32[n] = 32.f * b2[n * 16 + nn];

    for (int b = wave; b < BATCH; b += nW) {
        float sum4[4] = {0.f, 0.f, 0.f, 0.f};
#pragma unroll 1
        for (int t = 0; t < 2; ++t) {     // two 16-neighbor tiles per sample
            // ---- layer 0: A from x (K=16, padded to 32) ----
            const float* xr = x + (size_t)b * XPITCH + 32 + (t * 16 + nn) * 16 + kb;
            v16h a0;
#pragma unroll
            for (int e = 0; e < 8; ++e) { a0[e] = (_Float16)xr[e]; a0[e + 8] = (_Float16)0.f; }
#pragma unroll 1
            for (int n = 0; n < 8; ++n) {
                const _Float16* wp = w0T + (n * 16 + nn) * 32 + kb;
                v8f c = {};
                c = wmma16(a0, cmb(*(const v8h*)wp, *(const v8h*)(wp + 16)), c);
                float bias = bias0[n];
#pragma unroll
                for (int r = 0; r < 8; ++r) {
                    float v = fmaxf(c[r] + bias, 0.f);
                    A_[(mb + r) * 128 + n * 16 + nn] = (_Float16)v;
                }
            }
            // ---- layer 1: 128 -> 128 ----
#pragma unroll 1
            for (int n = 0; n < 8; ++n) {
                v8f c = {};
#pragma unroll
                for (int k = 0; k < 4; ++k) {
                    const _Float16* ap = A_ + nn * 128 + k * 32 + kb;
                    const _Float16* wp = w1T + (n * 16 + nn) * 128 + k * 32 + kb;
                    c = wmma16(cmb(*(const v8h*)ap, *(const v8h*)(ap + 16)),
                               cmb(*(const v8h*)wp, *(const v8h*)(wp + 16)), c);
                }
                float bias = bias1[n];
#pragma unroll
                for (int r = 0; r < 8; ++r) {
                    float v = fmaxf(c[r] + bias, 0.f);
                    B_[(mb + r) * 128 + n * 16 + nn] = (_Float16)v;
                }
            }
            // ---- layer 2: 128 -> 64, reduce over the 16 rows ----
#pragma unroll 1
            for (int n = 0; n < 4; ++n) {
                v8f c = {};
#pragma unroll
                for (int k = 0; k < 4; ++k) {
                    const _Float16* ap = B_ + nn * 128 + k * 32 + kb;
                    const _Float16* wp = w2T + (n * 16 + nn) * 128 + k * 32 + kb;
                    c = wmma16(cmb(*(const v8h*)ap, *(const v8h*)(ap + 16)),
                               cmb(*(const v8h*)wp, *(const v8h*)(wp + 16)), c);
                }
                float s = 0.f;
#pragma unroll
                for (int r = 0; r < 8; ++r) s += c[r];
                s += __shfl_down(s, 16, 32);   // add the M=8..15 half
                sum4[n] += s;                  // valid in lanes 0..15
            }
        }
        // ---- emit rho input row: [s_i | s_g | sum_h] in f16 ----
        _Float16* xrow = X + (size_t)b * 96;
        xrow[lane] = (_Float16)x[(size_t)b * XPITCH + lane];   // first 32 entries
        if (lane < 16) {
#pragma unroll
            for (int n = 0; n < 4; ++n)
                xrow[32 + n * 16 + lane] = (_Float16)(sum4[n] + bias2x32[n]);
        }
    }
}

// ------------------------------------------------------------------ rho -----
#define RHO_WAVES 4
__global__ void ds_rho(const float* __restrict__ rb0, const float* __restrict__ rb1,
                       const _Float16* __restrict__ ws, float* __restrict__ out) {
    __shared__ _Float16 ract[RHO_WAVES][16 * 256];
    const int lane = threadIdx.x & 31;
    const int wv   = threadIdx.x >> 5;
    const int wave = blockIdx.x * RHO_WAVES + wv;
    const int nW   = gridDim.x * RHO_WAVES;
    const _Float16* X    = ws + XBUF_OFF;
    const _Float16* rw0T = ws + RW0T_OFF;
    const _Float16* rw1T = ws + RW1T_OFF;
    const int nn = lane & 15;
    const int kb = (lane >> 4) * 8;
    const int mb = (lane >> 4) * 8;
    _Float16* A_ = ract[wv];

    float bias0[16];
#pragma unroll
    for (int n = 0; n < 16; ++n) bias0[n] = rb0[n * 16 + nn];
    const float bias1 = rb1[nn];

    for (int mt = wave; mt < BATCH / 16; mt += nW) {
        // ---- rho0: (16x96) @ (96x256), relu -> LDS ----
#pragma unroll 1
        for (int n = 0; n < 16; ++n) {
            v8f c = {};
#pragma unroll
            for (int k = 0; k < 3; ++k) {
                const _Float16* ap = X + (size_t)(mt * 16 + nn) * 96 + k * 32 + kb;
                const _Float16* wp = rw0T + (n * 16 + nn) * 96 + k * 32 + kb;
                c = wmma16(cmb(*(const v8h*)ap, *(const v8h*)(ap + 16)),
                           cmb(*(const v8h*)wp, *(const v8h*)(wp + 16)), c);
            }
            float bias = bias0[n];
#pragma unroll
            for (int r = 0; r < 8; ++r) {
                float v = fmaxf(c[r] + bias, 0.f);
                A_[(mb + r) * 256 + n * 16 + nn] = (_Float16)v;
            }
        }
        // ---- rho1: (16x256) @ (256x16) ----
        v8f c = {};
#pragma unroll
        for (int k = 0; k < 8; ++k) {
            const _Float16* ap = A_ + nn * 256 + k * 32 + kb;
            const _Float16* wp = rw1T + nn * 256 + k * 32 + kb;
            c = wmma16(cmb(*(const v8h*)ap, *(const v8h*)(ap + 16)),
                       cmb(*(const v8h*)wp, *(const v8h*)(wp + 16)), c);
        }
#pragma unroll
        for (int r = 0; r < 8; ++r)
            out[(size_t)(mt * 16 + mb + r) * 16 + nn] = c[r] + bias1;
    }
}

// --------------------------------------------------------------- launch -----
extern "C" void kernel_launch(void* const* d_in, const int* in_sizes, int n_in,
                              void* d_out, int out_size, void* d_ws, size_t ws_size,
                              hipStream_t stream) {
    const float* x   = (const float*)d_in[0];
    const float* w0  = (const float*)d_in[1];
    const float* b0  = (const float*)d_in[2];
    const float* w1  = (const float*)d_in[3];
    const float* b1  = (const float*)d_in[4];
    const float* w2  = (const float*)d_in[5];
    const float* b2  = (const float*)d_in[6];
    const float* rw0 = (const float*)d_in[7];
    const float* rb0 = (const float*)d_in[8];
    const float* rw1 = (const float*)d_in[9];
    const float* rb1 = (const float*)d_in[10];
    _Float16* ws = (_Float16*)d_ws;
    float* out = (float*)d_out;

    ds_prep<<<224, 256, 0, stream>>>(w0, w1, w2, rw0, rw1, ws);
    ds_phi<<<1024, 32 * PHI_WAVES, 0, stream>>>(x, b0, b1, b2, ws, ws + XBUF_OFF);
    ds_rho<<<256, 32 * RHO_WAVES, 0, stream>>>(rb0, rb1, ws, out);
}
